// Feedzai_60559038873895
// MI455X (gfx1250) — matrix operand order repaired and checked by reading, not compiled, not run on hardware
//
#include <hip/hip_runtime.h>

// MI455X / gfx1250, wave32. GRU scan, WMMA f16->f32 recurrent GEMMs.
// Latency-bound sequential scan (1024 dependent steps). Batch columns are
// independent -> 16 WGs x 16 batch rows (full M=16 WMMA tiles). 32MB state
// table stays L2-resident; all table deps are same-thread same-address, so
// relaxed AGENT-scope atomics suffice (WGP$ bypass) with deferred scatter.
// x-tiles are prefetched by the Tensor Data Mover into double-buffered LDS.

#define B_ 256
#define T_ 1024
#define F_ 18
#define U_ 32
#define N_ 1000
#define TB 16                 // batch rows per scan workgroup
#define NWG (B_ / TB)         // 16 workgroups
#define SCAN_THREADS 128      // 4 wave32s

typedef __attribute__((ext_vector_type(16))) _Float16 v16h;
typedef __attribute__((ext_vector_type(8)))  float    v8f;
typedef __attribute__((ext_vector_type(4)))  unsigned u32x4;
typedef __attribute__((ext_vector_type(8)))  int      i32x8;
typedef __attribute__((ext_vector_type(4)))  int      i32x4;

#if defined(__has_builtin)
#if __has_builtin(__builtin_amdgcn_tensor_load_to_lds) && \
    __has_builtin(__builtin_amdgcn_s_wait_tensorcnt)
#define HAVE_TDM 1
#endif
#endif
#ifndef HAVE_TDM
#define HAVE_TDM 0
#endif

static __device__ __forceinline__ float hsig(float v) {
    return fminf(fmaxf(0.2f * v + 0.5f, 0.0f), 1.0f);
}

#if HAVE_TDM
// 1-D TDM descriptor: move one contiguous 1KB x-tile (256 x 4B) to LDS.
// Group0: count=1 | lds_addr | global_addr(57b) | type=2 ("image").
// Group1: data_size=2(4B); tensor_dim0=256; tensor_dim1=1; tile_dim0=256;
//         tensor_dim0_stride=256; no padding/iterate/multicast.
// clang-23 form: (u32x4 g0, i32x8 g1, i32x4 g2, i32x4 g3, i32x8 pad, i32 cpol)
static __device__ __forceinline__ void tdm_load_xtile(const _Float16* gsrc,
                                                      unsigned lds_off) {
    unsigned long long ga = (unsigned long long)(uintptr_t)gsrc;
    u32x4 g0;
    g0[0] = 1u;                                   // count=1
    g0[1] = lds_off;                              // lds_addr (bytes)
    g0[2] = (unsigned)ga;                         // global_addr[31:0]
    g0[3] = (unsigned)((ga >> 32) & 0x1FFFFFFull) // global_addr[56:32]
          | (2u << 30);                           // type=2
    i32x8 g1;
    g1[0] = 0x20000;                              // data_size=2 -> 4 bytes
    g1[1] = (int)(256u << 16);                    // tensor_dim0=256 (lo16)
    g1[2] = (int)(1u << 16);                      // tensor_dim1=1 (lo16)
    g1[3] = (int)(256u << 16);                    // tile_dim0=256
    g1[4] = 0;                                    // tile_dim1/2 unused
    g1[5] = 256;                                  // tensor_dim0_stride
    g1[6] = 0;
    g1[7] = 0;
    i32x4 gz4 = {0, 0, 0, 0};
    i32x8 gz8 = {0, 0, 0, 0, 0, 0, 0, 0};
    __builtin_amdgcn_tensor_load_to_lds(g0, g1, gz4, gz4, gz8, 0);
}
#endif

// ---------------------------------------------------------------------------
// Prep: xT[t][b][0..31] (f16) = [id as float, x[b,t,1..17], zero pad].
// ---------------------------------------------------------------------------
__global__ void prep_xth(const float* __restrict__ x, const int* __restrict__ ids,
                         _Float16* __restrict__ xth) {
    int gid = blockIdx.x * blockDim.x + threadIdx.x;   // 0 .. B*T-1
    int b = gid & (B_ - 1);
    int t = gid >> 8;
    const float* xr = x + ((size_t)b * T_ + t) * F_;
    int idv = ids[(size_t)b * T_ + t];
    _Float16* o = xth + ((size_t)t * B_ + b) * 32;
    o[0] = (_Float16)(float)idv;
#pragma unroll
    for (int f = 1; f < F_; ++f) o[f] = (_Float16)xr[f];
#pragma unroll
    for (int f = F_; f < 32; ++f) o[f] = (_Float16)0.0f;
}

// ---------------------------------------------------------------------------
// Persistent scan kernel: one WG = 16 batch rows, loops over all T steps.
// Wc(64x96) f16: rows 0..31 = rkernel, 32..49 = kernel, pad 0.
// GEMM1 (waves 0-3): [h | x] @ Wc[:,0..63] + bias -> z, r
// GEMM2 (waves 0-1): [r*h | x] @ Wc[:,64..95] + bias -> hh
// ---------------------------------------------------------------------------
__global__ __launch_bounds__(SCAN_THREADS) void scan_gru(
    const int* __restrict__ ids,
    float* __restrict__ tab,              // workspace copy of table (N,B,U)
    const _Float16* __restrict__ xth,     // (T,B,32) f16
    const float* __restrict__ kern,       // (18,96)
    const float* __restrict__ rkern,      // (32,96)
    const float* __restrict__ bias,       // (96)
    const float* __restrict__ w1, const float* __restrict__ b1,
    const float* __restrict__ w2, const float* __restrict__ b2,
    float* __restrict__ out)
{
    __shared__ _Float16 sW[64][96];       // packed weights (built once)
    __shared__ float    sH[TB][U_];       // gathered h (f32, for gate math)
    __shared__ _Float16 sHf[TB][U_];      // gathered h (f16, WMMA operand)
    __shared__ _Float16 sXd[2][TB][32];   // x_t tiles (double buffered, TDM)
    __shared__ _Float16 sRH[TB][U_];      // r*h (f16 operand for GEMM2)
    __shared__ float    sHN[TB][U_];      // new hidden states
    __shared__ float    sD[TB][32];       // MLP hidden (epilogue)

    const int tid  = threadIdx.x;
    const int wave = tid >> 5;            // 0..3 (wave32)
    const int lane = tid & 31;
    const int n_   = lane & 15;           // tile col (B/C/D) or row (A)
    const int g    = lane >> 4;           // lane half-group
    const int b0   = blockIdx.x * TB;

    // ---- build packed f16 weights in LDS (once) ----
    for (int i = tid; i < 64 * 96; i += SCAN_THREADS) {
        int k = i / 96, n = i % 96;
        float v = (k < U_) ? rkern[k * 96 + n]
                           : ((k < U_ + F_) ? kern[(k - U_) * 96 + n] : 0.0f);
        sW[k][n] = (_Float16)v;
    }

#if HAVE_TDM
    if (wave == 0)                        // kick TDM for step 0 tile
        tdm_load_xtile(xth + (size_t)b0 * 32, (unsigned)(uintptr_t)&sXd[0][0][0]);
#endif
    __syncthreads();

    // ---- loop-invariant B fragments, held in VGPRs for all 1024 steps ----
    // B(32x16): lane group g holds K = e + 8*g + 8*(e>>3) (matches A).
    auto bfrag = [&](int ntile, int ktile) {
        v16h r;
#pragma unroll
        for (int e = 0; e < 16; ++e)
            r[e] = sW[ktile * 32 + e + 8 * g + 8 * (e >> 3)][ntile * 16 + n_];
        return r;
    };
    v16h Bzr0 = bfrag(wave, 0);           // z/r tile, K 0..31  (h rows)
    v16h Bzr1 = bfrag(wave, 1);           // z/r tile, K 32..63 (x rows)
    v16h Bh0 = {}, Bh1 = {};
    if (wave < 2) { Bh0 = bfrag(4 + wave, 0); Bh1 = bfrag(4 + wave, 1); }
    const float bzr = bias[wave * 16 + n_];
    const float bh  = (wave < 2) ? bias[64 + wave * 16 + n_] : 0.0f;

    const int gr = tid >> 3;              // gather row 0..15
    const int gc = (tid & 7) * 4;         // gather col 0,4,..,28
    size_t prev_base = 0;

    for (int t = 0; t < T_; ++t) {
        const int cur = t & 1;
        // ---- phase A: deferred scatter of hn(t-1), gather h(t), x mgmt ----
        // Table deps are same-thread same-address (identical col partition in
        // scatter & gather): HW keeps store->load same-address order; relaxed
        // AGENT atomics bypass the read-only WGP$ so L2 data is always seen.
        if (t != 0) {
#pragma unroll
            for (int j = 0; j < 4; ++j)
                __hip_atomic_store(&tab[prev_base + gc + j], sHN[gr][gc + j],
                                   __ATOMIC_RELAXED, __HIP_MEMORY_SCOPE_AGENT);
        }
        int    idv   = ids[(size_t)(b0 + gr) * T_ + t];
        size_t hbase = ((size_t)idv * B_ + (b0 + gr)) * U_;
        prev_base = hbase;
#pragma unroll
        for (int j = 0; j < 4; ++j) {
            float hv = __hip_atomic_load(&tab[hbase + gc + j],
                                         __ATOMIC_RELAXED, __HIP_MEMORY_SCOPE_AGENT);
            sH[gr][gc + j]  = hv;
            sHf[gr][gc + j] = (_Float16)hv;   // f16 mirror: no cvt on hot path
        }
#if HAVE_TDM
        if (wave == 0) {
            __builtin_amdgcn_s_wait_tensorcnt(0);      // tile(t) landed
            if (t + 1 < T_)                            // prefetch tile(t+1)
                tdm_load_xtile(xth + ((size_t)(t + 1) * B_ + b0) * 32,
                               (unsigned)(uintptr_t)&sXd[(t + 1) & 1][0][0]);
        }
#else
        {
            const _Float16* xr = xth + ((size_t)t * B_ + (b0 + gr)) * 32;
#pragma unroll
            for (int j = 0; j < 4; ++j) sXd[cur][gr][gc + j] = xr[gc + j];
            if (t + 1 < T_)
                __builtin_prefetch(xth + ((size_t)(t + 1) * B_ + (b0 + gr)) * 32, 0, 0);
        }
#endif
        __syncthreads();

        // ---- A fragments (16-bit A 16x32: k = e + 8g + 8*(e>>3)) ----
        v16h aH, aX;
#pragma unroll
        for (int e = 0; e < 16; ++e) {
            int k = e + 8 * g + 8 * (e >> 3);
            aH[e] = sHf[n_][k];
            aX[e] = sXd[cur][n_][k];
        }

        // ---- phase B: GEMM1 -> z (waves 0,1) / r (waves 2,3) ----
        v8f acc;
#pragma unroll
        for (int v = 0; v < 8; ++v) acc[v] = bzr;
        acc = __builtin_amdgcn_wmma_f32_16x16x32_f16(false, aH, false, Bzr0,
                                                     (short)0, acc, false, false);
        acc = __builtin_amdgcn_wmma_f32_16x16x32_f16(false, aX, false, Bzr1,
                                                     (short)0, acc, false, false);
        float zreg[8] = {};
        if (wave >= 2) {                  // r tiles -> write r*h operand
            int u = (wave - 2) * 16 + n_;
#pragma unroll
            for (int v = 0; v < 8; ++v) {
                int m = v + 8 * g;
                float r = hsig(acc[v]);
                sRH[m][u] = (_Float16)(r * sH[m][u]);
            }
        } else {                          // z tiles -> keep in registers
#pragma unroll
            for (int v = 0; v < 8; ++v) zreg[v] = hsig(acc[v]);
        }
        __syncthreads();

        // ---- phase C: GEMM2 -> hh, hn (waves 0,1) ----
        if (wave < 2) {
            v16h aRH;
#pragma unroll
            for (int e = 0; e < 16; ++e) {
                int k = e + 8 * g + 8 * (e >> 3);
                aRH[e] = sRH[n_][k];
            }
            v8f acc2;
#pragma unroll
            for (int v = 0; v < 8; ++v) acc2[v] = bh;
            acc2 = __builtin_amdgcn_wmma_f32_16x16x32_f16(false, aRH, false, Bh0,
                                                          (short)0, acc2, false, false);
            acc2 = __builtin_amdgcn_wmma_f32_16x16x32_f16(false, aX, false, Bh1,
                                                          (short)0, acc2, false, false);
            int u = wave * 16 + n_;
#pragma unroll
            for (int v = 0; v < 8; ++v) {
                int m = v + 8 * g;
                float hh = tanhf(acc2[v]);
                float h  = sH[m][u];
                float z  = zreg[v];
                sHN[m][u] = z * h + (1.0f - z) * hh;
            }
        }
        __syncthreads();
        // last step's hn is never scattered: table is dead after the scan.
    }

    // ---- epilogue MLP on final hn (runs once; VALU is fine) ----
    {
        int m = tid >> 3, j0 = (tid & 7) * 4;
#pragma unroll
        for (int jj = 0; jj < 4; ++jj) {
            int jc = j0 + jj;
            float s = b1[jc];
#pragma unroll
            for (int u = 0; u < U_; ++u) s += sHN[m][u] * w1[u * 32 + jc];
            sD[m][jc] = fmaxf(s, 0.0f);
        }
        __syncthreads();
        if (tid < TB) {
            float s = b2[0];
#pragma unroll
            for (int j = 0; j < 32; ++j) s += sD[tid][j] * w2[j];
            out[b0 + tid] = 1.0f / (1.0f + expf(-s));
        }
    }
}

// ---------------------------------------------------------------------------
extern "C" void kernel_launch(void* const* d_in, const int* in_sizes, int n_in,
                              void* d_out, int out_size, void* d_ws, size_t ws_size,
                              hipStream_t stream) {
    (void)in_sizes; (void)n_in; (void)out_size; (void)ws_size;
    const float* x     = (const float*)d_in[0];
    const int*   ids   = (const int*)  d_in[1];
    const float* table = (const float*)d_in[2];
    const float* kern  = (const float*)d_in[3];
    const float* rkern = (const float*)d_in[4];
    const float* bias  = (const float*)d_in[5];
    const float* w1    = (const float*)d_in[6];
    const float* b1    = (const float*)d_in[7];
    const float* w2    = (const float*)d_in[8];
    const float* b2    = (const float*)d_in[9];
    float* out = (float*)d_out;

    // ws: [0,16MB) xT f16 ; [16MB, 16MB+32MB) mutable table copy
    const size_t xth_bytes = (size_t)T_ * B_ * 32 * 2;             // 16.78 MB
    const size_t tab_bytes = (size_t)N_ * B_ * U_ * sizeof(float); // 32.77 MB
    _Float16* xth = (_Float16*)d_ws;
    float*    tab = (float*)((char*)d_ws + xth_bytes);

    // never mutate the input table: fresh copy every launch (deterministic)
    (void)hipMemcpyAsync(tab, table, tab_bytes, hipMemcpyDeviceToDevice, stream);
    prep_xth<<<(B_ * T_) / 256, 256, 0, stream>>>(x, ids, xth);
    scan_gru<<<NWG, SCAN_THREADS, 0, stream>>>(ids, tab, xth, kern, rkern, bias,
                                               w1, b1, w2, b2, out);
}
